// LSTMSimple_18485539242756
// MI455X (gfx1250) — compile-verified, matrix operand order checked by reading, not run on hardware
//
#include <hip/hip_runtime.h>
#include <hip/hip_fp16.h>
#include <stdint.h>

#define B_   64
#define T_   1024
#define D_   128
#define H_   512
#define G4_  2048            // 4*H
#define M_   (B_ * T_)       // 65536 rows

typedef _Float16 h16;
typedef _Float16 v16h __attribute__((ext_vector_type(16)));
typedef float    v8f  __attribute__((ext_vector_type(8)));
typedef float    v4f  __attribute__((ext_vector_type(4)));

// ---------------- workspace layout (bytes) ----------------
static constexpr size_t XG_OFF    = 0;                         // f16 (M_ x 2048)  256MB
static constexpr size_t H1_OFF    = XG_OFF    + 268435456ULL;  // f16 (M_ x 512)   64MB (reused for h2)
static constexpr size_t H1N_OFF   = H1_OFF    + 67108864ULL;   // f16 (M_ x 512)   64MB
static constexpr size_t XF16_OFF  = H1N_OFF   + 67108864ULL;   // f16 (M_ x 128)   16MB
static constexpr size_t WXT1_OFF  = XF16_OFF  + 16777216ULL;   // f16 (2048 x 128)
static constexpr size_t WHT1_OFF  = WXT1_OFF  + 524288ULL;     // f16 (2048 x 512)
static constexpr size_t WXT2_OFF  = WHT1_OFF  + 2097152ULL;    // f16 (2048 x 512)
static constexpr size_t WHT2_OFF  = WXT2_OFF  + 2097152ULL;    // f16 (2048 x 512)
static constexpr size_t MEAN_OFF  = WHT2_OFF  + 2097152ULL;    // f32 x 512
static constexpr size_t RSTD_OFF  = MEAN_OFF  + 2048ULL;
static constexpr size_t HLAST_OFF = RSTD_OFF  + 2048ULL;       // f32 (64 x 512)

// ---------------- TDM availability ----------------
#if defined(__has_builtin)
#  if __has_builtin(__builtin_amdgcn_tensor_load_to_lds) && __has_builtin(__builtin_amdgcn_s_wait_tensorcnt)
#    define HAVE_TDM 1
#  endif
#endif
#ifndef HAVE_TDM
#  define HAVE_TDM 0
#endif

#if HAVE_TDM
typedef unsigned int u32x4 __attribute__((ext_vector_type(4)));
typedef int          i32x8 __attribute__((ext_vector_type(8)));
typedef int          i32x4 __attribute__((ext_vector_type(4)));

// 2D tile load: tile_d0 elements (2B) per row, tile_d1 rows, row stride in elements.
// D# packing per CDNA5 ISA (groups 0/1; 2D so groups 2/3 zero).
// This toolchain exposes the 6-arg builtin: (g0, g1, g2, g3, g4, cpol).
__device__ __forceinline__ void tdm_load_2d(unsigned lds_byte_off, const void* gptr,
                                            unsigned tile_d0, unsigned tile_d1,
                                            unsigned long long row_stride_elems) {
  unsigned long long ga = (unsigned long long)gptr;
  u32x4 g0;
  g0[0] = 1u;                                               // count=1, user descriptor
  g0[1] = lds_byte_off;                                     // lds_addr
  g0[2] = (unsigned)(ga & 0xFFFFFFFFu);                     // global_addr[31:0]
  g0[3] = (unsigned)((ga >> 32) & 0x01FFFFFFu) | (2u << 30);// global_addr[56:32] | type=2
  i32x8 g1;
  g1[0] = (int)(1u << 16);                                  // wg_mask=0, data_size=1 (2B)
  g1[1] = (int)((tile_d0 & 0xFFFFu) << 16);                 // tensor_dim0[15:0]
  g1[2] = (int)((tile_d0 >> 16) | ((tile_d1 & 0xFFFFu) << 16));   // tdim0 hi | tensor_dim1 lo
  g1[3] = (int)(((tile_d1 >> 16) & 0xFFFFu) | ((tile_d0 & 0xFFFFu) << 16)); // tdim1 hi | tile_dim0
  g1[4] = (int)(tile_d1 & 0xFFFFu);                         // tile_dim1 | tile_dim2=0
  g1[5] = (int)(unsigned)(row_stride_elems & 0xFFFFFFFFull);// tensor_dim0_stride[31:0]
  g1[6] = (int)(unsigned)((row_stride_elems >> 32) & 0xFFFFull); // stride[47:32] | tdim1_stride lo = 0
  g1[7] = 0;
  i32x4 z4 = {0, 0, 0, 0};
  i32x8 z8 = {0, 0, 0, 0, 0, 0, 0, 0};
  __builtin_amdgcn_tensor_load_to_lds(g0, g1, z4, z4, z8, 0);
}
#endif

// ---------------- helpers ----------------
__device__ __forceinline__ v16h load_frag16(const h16* p) {
  // 8-half run at p, second run at p+16 (matches 16-bit A/B K-striping)
  union { v16h v; v4f q[2]; } u;
  u.q[0] = *(const v4f*)(p);
  u.q[1] = *(const v4f*)(p + 16);
  return u.v;
}
__device__ __forceinline__ float sigf(float x) { return 1.0f / (1.0f + __expf(-x)); }
__device__ __forceinline__ float tanh_fast(float x) {
  float t = __expf(-2.0f * fabsf(x));
  float r = (1.0f - t) / (1.0f + t);
  return copysignf(r, x);
}

// ---------------- conversion kernels ----------------
__global__ void cvt_f32_to_f16(const float* __restrict__ in, h16* __restrict__ out, int n) {
  int i = blockIdx.x * blockDim.x + threadIdx.x;
  if (i < n) out[i] = (h16)in[i];
}
// W (K rows x N cols, f32 row-major) -> WT (N x K, f16)
__global__ void transpose_cvt(const float* __restrict__ in, h16* __restrict__ out, int K, int N) {
  int i = blockIdx.x * blockDim.x + threadIdx.x;
  if (i >= N * K) return;
  int n = i / K, k = i - n * K;
  out[i] = (h16)in[(size_t)k * N + n];
}

// ---------------- WMMA GEMM: out(M x 2048) = A(M x K) * WT^T + bias ----------------
// grid: (M/64, 2048/64), block: 128 (4 waves). Each wave: 16 rows x 64 cols (4 tiles).
__global__ __launch_bounds__(128) void gemm_bias_wmma(
    const h16* __restrict__ A, const h16* __restrict__ WT,
    const float* __restrict__ bias, h16* __restrict__ out, int M, int K) {
  const int wave = threadIdx.x >> 5;
  const int lane = threadIdx.x & 31;
  const int lrow = lane & 15;
  const int hs   = lane >> 4;               // half-select: K run offset
  const int row0 = blockIdx.x * 64 + wave * 16;
  const int col0 = blockIdx.y * 64;

  v8f acc[4] = {v8f{0}, v8f{0}, v8f{0}, v8f{0}};
  const h16* arow = A + (size_t)(row0 + lrow) * K;
  for (int kb = 0; kb < K; kb += 32) {
    v16h af = load_frag16(arow + kb + hs * 8);
    if (kb + 128 < K) __builtin_prefetch(arow + kb + 128, 0, 1);  // global_prefetch_b8
    #pragma unroll
    for (int g = 0; g < 4; ++g) {
      const h16* brow = WT + (size_t)(col0 + g * 16 + lrow) * K + kb + hs * 8;
      v16h bf = load_frag16(brow);
      acc[g] = __builtin_amdgcn_wmma_f32_16x16x32_f16(false, af, false, bf,
                                                      (short)0, acc[g], false, false);
    }
  }
  #pragma unroll
  for (int g = 0; g < 4; ++g) {
    int n = col0 + g * 16 + lrow;
    float bv = bias[n];
    #pragma unroll
    for (int e = 0; e < 8; ++e) {
      int r = row0 + e + hs * 8;
      out[(size_t)r * G4_ + n] = (h16)(acc[g][e] + bv);
    }
  }
}

// ---------------- persistent LSTM scan, batch-partitioned ----------------
// grid: 4 blocks x 512 threads (16 waves). Block r owns batch rows [16r, 16r+16) and
// computes ALL 2048 gate columns (wave w owns h cols [32w, 32w+32) + its 4 gate groups).
// h ping-pongs in LDS (workgroup barrier only, no global sync). The next step's xg
// slice (16 x 2048 f16) is fetched by the Tensor Data Mover into a double-buffered
// LDS stage, overlapped with the current step's WMMAs.
struct ScanSmem {
  h16 xgs[2][16][G4_];   // 2 x 64KB xg stage (TDM destination)
  h16 hb[2][16][H_];     // 2 x 16KB h ping-pong
};

__global__ __launch_bounds__(512, 1) void lstm_scan_wmma(
    const h16* __restrict__ xg,      // (B, T, 2048) f16, bias pre-added
    const h16* __restrict__ wht,     // (2048 x 512) f16 N-major
    h16* __restrict__ hout) {        // (B, T, 512) f16
  __shared__ ScanSmem sm;            // 160KB (<= 320KB WGP LDS)

  const int b0   = blockIdx.x * 16;
  const int wave = threadIdx.x >> 5;
  const int lane = threadIdx.x & 31;
  const int lrow = lane & 15;
  const int hs   = lane >> 4;
  const int j0   = wave * 32;        // this wave's h columns [j0, j0+32)

  // h_{-1} = 0
  for (int i = threadIdx.x; i < 16 * H_; i += blockDim.x) (&sm.hb[0][0][0])[i] = (h16)0.f;

  // stage xg slice for t=0 into buffer 0
#if HAVE_TDM
  if (threadIdx.x == 0) {
    tdm_load_2d(0u, xg + (size_t)b0 * T_ * G4_, G4_, 16, (unsigned long long)T_ * G4_);
    __builtin_amdgcn_s_wait_tensorcnt(0);
  }
#else
  for (int ci = threadIdx.x; ci < (16 * G4_) / 8; ci += blockDim.x) {
    int r = ci / (G4_ / 8), cc = ci - r * (G4_ / 8);
    *(v4f*)&sm.xgs[0][r][cc * 8] = *(const v4f*)(xg + ((size_t)(b0 + r) * T_) * G4_ + cc * 8);
  }
#endif
  __syncthreads();

  v8f cst[2] = {v8f{0}, v8f{0}};     // cell state, VGPR-resident across all timesteps

  for (int t = 0; t < T_; ++t) {
    const int cb = t & 1, nb = cb ^ 1;

    // kick off async TDM fetch of next step's xg slice into the other stage buffer
    if (t + 1 < T_) {
#if HAVE_TDM
      if (threadIdx.x == 0)
        tdm_load_2d((unsigned)(nb * 65536), xg + ((size_t)b0 * T_ + t + 1) * G4_,
                    G4_, 16, (unsigned long long)T_ * G4_);
#else
      for (int ci = threadIdx.x; ci < (16 * G4_) / 8; ci += blockDim.x) {
        int r = ci / (G4_ / 8), cc = ci - r * (G4_ / 8);
        *(v4f*)&sm.xgs[nb][r][cc * 8] =
            *(const v4f*)(xg + ((size_t)(b0 + r) * T_ + t + 1) * G4_ + cc * 8);
      }
#endif
    }

    // accumulators start at xg slice (bias already folded in)
    v8f acc[4][2];
    #pragma unroll
    for (int g = 0; g < 4; ++g)
      #pragma unroll
      for (int jt = 0; jt < 2; ++jt) {
        int n = g * H_ + j0 + jt * 16 + lrow;
        #pragma unroll
        for (int e = 0; e < 8; ++e)
          acc[g][jt][e] = (float)sm.xgs[cb][e + hs * 8][n];
      }

    // gates += h_prev(16x512) @ Wh(512 x this wave's 128 gate cols)
    for (int kb = 0; kb < H_; kb += 32) {
      v16h af = load_frag16(&sm.hb[cb][lrow][kb + hs * 8]);
      #pragma unroll
      for (int g = 0; g < 4; ++g)
        #pragma unroll
        for (int jt = 0; jt < 2; ++jt) {
          const h16* brow = wht + (size_t)(g * H_ + j0 + jt * 16 + lrow) * H_ + kb + hs * 8;
          v16h bf = load_frag16(brow);
          acc[g][jt] = __builtin_amdgcn_wmma_f32_16x16x32_f16(false, af, false, bf,
                                                              (short)0, acc[g][jt], false, false);
        }
    }

    // elementwise LSTM cell (wave-local: i/f/g/o tiles share (row, j) layout)
    #pragma unroll
    for (int jt = 0; jt < 2; ++jt)
      #pragma unroll
      for (int e = 0; e < 8; ++e) {
        float iv = sigf(acc[0][jt][e]);
        float fv = sigf(acc[1][jt][e]);
        float gv = tanh_fast(acc[2][jt][e]);
        float ov = sigf(acc[3][jt][e]);
        float c  = fv * cst[jt][e] + iv * gv;
        cst[jt][e] = c;
        float h  = ov * tanh_fast(c);
        int row  = e + hs * 8;
        int j    = j0 + jt * 16 + lrow;
        h16 hv   = (h16)h;
        sm.hb[nb][row][j] = hv;
        hout[((size_t)(b0 + row) * T_ + t) * H_ + j] = hv;
      }

#if HAVE_TDM
    if (threadIdx.x == 0) __builtin_amdgcn_s_wait_tensorcnt(0);
#endif
    __syncthreads();   // h_next + next xg stage ready for step t+1
  }
}

// ---------------- batchnorm ----------------
__global__ void bn_stats(const h16* __restrict__ h, float* __restrict__ mean, float* __restrict__ rstd) {
  __shared__ float ss[256], sq[256];
  int j = blockIdx.x;                 // channel
  float s = 0.f, q = 0.f;
  for (int r = threadIdx.x; r < M_; r += 256) {
    float v = (float)h[(size_t)r * H_ + j];
    s += v; q += v * v;
  }
  ss[threadIdx.x] = s; sq[threadIdx.x] = q;
  __syncthreads();
  for (int st = 128; st > 0; st >>= 1) {
    if (threadIdx.x < st) { ss[threadIdx.x] += ss[threadIdx.x + st]; sq[threadIdx.x] += sq[threadIdx.x + st]; }
    __syncthreads();
  }
  if (threadIdx.x == 0) {
    float m = ss[0] / (float)M_;
    float v = sq[0] / (float)M_ - m * m;
    mean[j] = m;
    rstd[j] = rsqrtf(v + 1e-5f);
  }
}

__global__ void bn_apply(const h16* __restrict__ h, const float* __restrict__ mean,
                         const float* __restrict__ rstd, const float* __restrict__ sc,
                         const float* __restrict__ bi, h16* __restrict__ out) {
  size_t i = (size_t)blockIdx.x * blockDim.x + threadIdx.x;
  if (i >= (size_t)M_ * H_) return;
  int j = (int)(i & (H_ - 1));
  float v = (float)h[i];
  out[i] = (h16)((v - mean[j]) * rstd[j] * sc[j] + bi[j]);
}

__global__ void bn_last(const h16* __restrict__ h, const float* __restrict__ mean,
                        const float* __restrict__ rstd, const float* __restrict__ sc,
                        const float* __restrict__ bi, float* __restrict__ hlast) {
  int i = blockIdx.x * blockDim.x + threadIdx.x;     // 64*512
  if (i >= B_ * H_) return;
  int b = i >> 9, j = i & (H_ - 1);
  float v = (float)h[((size_t)b * T_ + (T_ - 1)) * H_ + j];
  hlast[i] = (v - mean[j]) * rstd[j] * sc[j] + bi[j];
}

// ---------------- MLP head ----------------
__global__ void head_kernel(const float* __restrict__ hlast, const float* __restrict__ Wd1,
                            const float* __restrict__ bd1, const float* __restrict__ Wd2,
                            const float* __restrict__ bd2, float* __restrict__ out) {
  int b = threadIdx.x;   // 64 threads
  if (b >= B_) return;
  float hid[16];
  #pragma unroll
  for (int j = 0; j < 16; ++j) hid[j] = bd1[j];
  for (int c = 0; c < H_; ++c) {
    float hv = hlast[b * H_ + c];
    #pragma unroll
    for (int j = 0; j < 16; ++j) hid[j] += hv * Wd1[c * 16 + j];
  }
  float o = bd2[0];
  #pragma unroll
  for (int j = 0; j < 16; ++j) o += fmaxf(hid[j], 0.f) * Wd2[j];
  out[b] = o;
}

// ---------------- launch ----------------
extern "C" void kernel_launch(void* const* d_in, const int* in_sizes, int n_in,
                              void* d_out, int out_size, void* d_ws, size_t ws_size,
                              hipStream_t stream) {
  const float* x      = (const float*)d_in[0];
  const float* Wx1    = (const float*)d_in[1];
  const float* Wh1    = (const float*)d_in[2];
  const float* b1     = (const float*)d_in[3];
  const float* bn1s   = (const float*)d_in[4];
  const float* bn1b   = (const float*)d_in[5];
  const float* Wx2    = (const float*)d_in[6];
  const float* Wh2    = (const float*)d_in[7];
  const float* b2     = (const float*)d_in[8];
  const float* bn2s   = (const float*)d_in[9];
  const float* bn2b   = (const float*)d_in[10];
  const float* Wd1    = (const float*)d_in[11];
  const float* bd1    = (const float*)d_in[12];
  const float* Wd2    = (const float*)d_in[13];
  const float* bd2    = (const float*)d_in[14];

  uint8_t* ws = (uint8_t*)d_ws;
  h16*   xg    = (h16*)(ws + XG_OFF);
  h16*   h1    = (h16*)(ws + H1_OFF);     // h2 reuses this after layer 1
  h16*   h1n   = (h16*)(ws + H1N_OFF);
  h16*   xf16  = (h16*)(ws + XF16_OFF);
  h16*   wxt1  = (h16*)(ws + WXT1_OFF);
  h16*   wht1  = (h16*)(ws + WHT1_OFF);
  h16*   wxt2  = (h16*)(ws + WXT2_OFF);
  h16*   wht2  = (h16*)(ws + WHT2_OFF);
  float* mean  = (float*)(ws + MEAN_OFF);
  float* rstd  = (float*)(ws + RSTD_OFF);
  float* hlast = (float*)(ws + HLAST_OFF);

  // precision conversions / weight transposes
  cvt_f32_to_f16<<<(M_ * D_) / 256, 256, 0, stream>>>(x, xf16, M_ * D_);
  transpose_cvt<<<(G4_ * D_) / 256, 256, 0, stream>>>(Wx1, wxt1, D_, G4_);
  transpose_cvt<<<(G4_ * H_) / 256, 256, 0, stream>>>(Wh1, wht1, H_, G4_);
  transpose_cvt<<<(G4_ * H_) / 256, 256, 0, stream>>>(Wx2, wxt2, H_, G4_);
  transpose_cvt<<<(G4_ * H_) / 256, 256, 0, stream>>>(Wh2, wht2, H_, G4_);

  dim3 ggrid(M_ / 64, G4_ / 64);

  // ---- layer 1 ----
  gemm_bias_wmma<<<ggrid, 128, 0, stream>>>(xf16, wxt1, b1, xg, M_, D_);
  lstm_scan_wmma<<<B_ / 16, 512, 0, stream>>>(xg, wht1, h1);
  bn_stats<<<H_, 256, 0, stream>>>(h1, mean, rstd);
  bn_apply<<<(M_ * H_) / 256, 256, 0, stream>>>(h1, mean, rstd, bn1s, bn1b, h1n);

  // ---- layer 2 ----
  gemm_bias_wmma<<<ggrid, 128, 0, stream>>>(h1n, wxt2, b2, xg, M_, H_);
  lstm_scan_wmma<<<B_ / 16, 512, 0, stream>>>(xg, wht2, h1 /* = h2 */);
  bn_stats<<<H_, 256, 0, stream>>>(h1, mean, rstd);
  bn_last<<<(B_ * H_) / 256, 256, 0, stream>>>(h1, mean, rstd, bn2s, bn2b, hlast);

  // ---- head ----
  head_kernel<<<1, 64, 0, stream>>>(hlast, Wd1, bd1, Wd2, bd2, (float*)d_out);
}
